// BigBirdPegasusBlockSparseAttention_64544768524419
// MI455X (gfx1250) — matrix operand order repaired and checked by reading, not compiled, not run on hardware
//
#include <hip/hip_runtime.h>
#include <hip/hip_bf16.h>
#include <stdint.h>

// Problem constants (match reference)
#define B_    2
#define S_    4096
#define H_    768
#define NH_   12
#define BLK_  64
#define NB_   64
#define D_    64
#define R_    3
#define PEN   (-10000.0f)

typedef __attribute__((ext_vector_type(16))) __bf16 v16bf;
typedef __attribute__((ext_vector_type(8)))  float  v8f;

struct B32A { uint4 a, b; };                 // 32 bytes

__device__ __forceinline__ unsigned short f2bf(float f) {
  unsigned u = __float_as_uint(f);
  u += 0x7FFFu + ((u >> 16) & 1u);           // round-to-nearest-even
  return (unsigned short)(u >> 16);
}

// Build a 16-element bf16 fragment from two 16-byte chunks.
__device__ __forceinline__ v16bf ld_frag(const unsigned short* p0, const unsigned short* p1) {
  B32A t;
  t.a = *(const uint4*)p0;
  t.b = *(const uint4*)p1;
  return __builtin_bit_cast(v16bf, t);
}

// ---------------------------------------------------------------------------
// Kernel 0: one-time conversions.
//   X  (B*S x H) f32            -> Xb  bf16 (row-major, unchanged layout)
//   Wq/Wk/Wv (H x H) f32        -> WTb bf16 TRANSPOSED: WTb[mat][n][k]
// ---------------------------------------------------------------------------
__global__ __launch_bounds__(256) void convert_kernel(
    const float* __restrict__ X,
    const float* __restrict__ Wq, const float* __restrict__ Wk, const float* __restrict__ Wv,
    unsigned short* __restrict__ Xb, unsigned short* __restrict__ WTb)
{
  const size_t NX = (size_t)B_ * S_ * H_;          // 6,291,456
  const size_t NW = (size_t)3 * H_ * H_;           // 1,769,472
  size_t i = (size_t)blockIdx.x * blockDim.x + threadIdx.x;
  if (i < NX) {
    Xb[i] = f2bf(X[i]);
    return;
  }
  size_t j = i - NX;
  if (j >= NW) return;
  int mat = (int)(j / (H_ * H_));
  int r   = (int)(j % (H_ * H_));
  int k = r / H_, n = r % H_;                      // coalesced read along n
  const float* W = (mat == 0) ? Wq : (mat == 1) ? Wk : Wv;
  WTb[(size_t)mat * H_ * H_ + (size_t)n * H_ + k] = f2bf(W[r]);
}

// ---------------------------------------------------------------------------
// Kernel 1: Q/K/V projection GEMM, pure bf16 WMMA (no LDS, no conversions).
//   (B*S x H) x (H x H) -> bf16 [b][h][s][d]
// ---------------------------------------------------------------------------
__global__ __launch_bounds__(128) void qkv_proj_kernel(
    const unsigned short* __restrict__ Xb, const unsigned short* __restrict__ WTb,
    unsigned short* __restrict__ Qw, unsigned short* __restrict__ Kw, unsigned short* __restrict__ Vw)
{
  const int bm  = blockIdx.x;        // 64-row tile of B*S
  const int bn  = blockIdx.y;        // head (64-col tile of H)
  const int mat = blockIdx.z;        // 0=Q 1=K 2=V
  unsigned short* O = (mat == 0) ? Qw : (mat == 1) ? Kw : Vw;
  const unsigned short* Wt = WTb + (size_t)mat * H_ * H_;   // [n][k] bf16

  const int tid  = threadIdx.x;
  const int w    = tid >> 5;
  const int lane = tid & 31;
  const int l16  = lane & 15;
  const int hi   = lane >> 4;
  const int rowA = bm * 64 + w * 16 + l16;

  const unsigned short* xr = Xb + (size_t)rowA * H_;

  v8f zero = {};
  v8f acc[4] = {zero, zero, zero, zero};

  for (int k0 = 0; k0 < H_; k0 += 32) {
    // A fragment: lanes hold K = hi*8..+7 and +16..+23 (ISA 16-bit A layout)
    const unsigned short* ap = xr + k0 + hi * 8;
    v16bf aF = ld_frag(ap, ap + 16);

    #pragma unroll
    for (int t = 0; t < 4; ++t) {
      // B fragment: lane col n = l16 (+16t); K = k0 + hi*8..+7 and +16..+23 in WT[n][k]
      const unsigned short* bp = Wt + (size_t)(bn * 64 + t * 16 + l16) * H_ + k0 + hi * 8;
      v16bf bF = ld_frag(bp, bp + 16);
      acc[t] = __builtin_amdgcn_wmma_f32_16x16x32_bf16(
          false, aF, false, bF, (short)0, acc[t], false, false);
    }
  }

  // C/D layout: lane holds col n = l16 (+16t), rows m = hi*8 + v
  #pragma unroll
  for (int v = 0; v < 8; ++v) {
    int rg = bm * 64 + w * 16 + hi * 8 + v;
    int bb = rg >> 12, ss = rg & (S_ - 1);
    size_t ob = ((size_t)(bb * NH_ + bn) * S_ + ss) * D_;
    #pragma unroll
    for (int t = 0; t < 4; ++t) O[ob + t * 16 + l16] = f2bf(acc[t][v]);
  }
}

// ---------------------------------------------------------------------------
// Key-block list for a given q-block (uniform per workgroup).
// mode 0 = to_mask, 1 = band_mask (rel=0..2), 2 = rand (fb*tb)
// ---------------------------------------------------------------------------
__device__ __forceinline__ void key_block_for(
    int qb, int it, int b, int h, const long long* __restrict__ rand_attn,
    int& kb, int& mode, int& rel)
{
  rel = 0;
  if (qb == 0 || qb == NB_ - 1) { kb = it; mode = 0; return; }
  const int l = qb - 1;
  const long long* ra = rand_attn + (((size_t)b * NH_ + h) * (NB_ - 2) + l) * R_;
  if (qb == 1) {
    if (it < 3)       { kb = it;        mode = 0; }
    else if (it == 3) { kb = NB_ - 1;   mode = 0; }
    else              { kb = (int)ra[it - 4]; mode = 2; }
  } else if (qb == NB_ - 2) {
    if (it == 0)      { kb = 0;            mode = 0; }
    else if (it < 4)  { kb = NB_ - 4 + it; mode = 0; }   // 61,62,63
    else              { kb = (int)ra[it - 4]; mode = 2; }
  } else {
    if (it == 0)      { kb = 0;           mode = 0; }
    else if (it < 4)  { kb = qb - 2 + it; mode = 1; rel = it - 1; }
    else if (it < 7)  { kb = (int)ra[it - 4]; mode = 2; }
    else              { kb = NB_ - 1;     mode = 0; }
  }
}

// ---------------------------------------------------------------------------
// Kernel 2: flash-style block-sparse attention, one WG per (b,h,qblock)
// ---------------------------------------------------------------------------
__global__ __launch_bounds__(128) void bigbird_attn_kernel(
    const unsigned short* __restrict__ Qw, const unsigned short* __restrict__ Kw,
    const unsigned short* __restrict__ Vw,
    const float* __restrict__ band_mask, const float* __restrict__ from_mask,
    const float* __restrict__ to_mask,   const float* __restrict__ from_blocked,
    const float* __restrict__ to_blocked, const long long* __restrict__ rand_attn,
    float* __restrict__ out)
{
  const int qb = blockIdx.x, h = blockIdx.y, b = blockIdx.z;
  const int tid  = threadIdx.x;
  const int w    = tid >> 5;
  const int lane = tid & 31;
  const int l16  = lane & 15;
  const int hi   = lane >> 4;

  __shared__ alignas(16) unsigned short VT[64 * 72];       // V^T [d][key], padded
  __shared__ alignas(16) unsigned short Pl[4][16 * 72];    // per-wave P tile [row][key]

  const size_t headoff = ((size_t)b * NH_ + h) * (size_t)S_ * D_;
  const unsigned short* Qb = Qw + headoff;
  const unsigned short* Kb = Kw + headoff;
  const unsigned short* Vb = Vw + headoff;

  // Q A-fragments for this wave's 16 query rows (2 k-steps over d)
  const int qrowA = qb * 64 + w * 16 + l16;
  v16bf qA[2];
  #pragma unroll
  for (int ks = 0; ks < 2; ++ks) {
    const unsigned short* p = Qb + (size_t)qrowA * D_ + ks * 32 + hi * 8;
    qA[ks] = ld_frag(p, p + 16);
  }

  float fbv[8];
  #pragma unroll
  for (int v = 0; v < 8; ++v)
    fbv[v] = from_blocked[((size_t)b * NB_ + qb) * BLK_ + w * 16 + hi * 8 + v];

  float mrow[8], lrow[8];
  #pragma unroll
  for (int v = 0; v < 8; ++v) { mrow[v] = -1e30f; lrow[v] = 0.f; }
  v8f zero = {};
  v8f accO[4] = {zero, zero, zero, zero};

  const int nkb = (qb == 0 || qb == NB_ - 1) ? NB_
                : ((qb == 1 || qb == NB_ - 2) ? 7 : 8);

  for (int it = 0; it < nkb; ++it) {
    int kb, mode, rel;
    key_block_for(qb, it, b, h, rand_attn, kb, mode, rel);

    __syncthreads();
    // Stage V^T: b128 global loads (8 bf16 per chunk), transposing b16 DS stores.
    for (int e = tid; e < 64 * 8; e += 128) {
      int key = e >> 3;              // 0..63
      int dc  = (e & 7) * 8;         // 0,8,..,56
      union { uint4 q; unsigned short s[8]; } u;
      u.q = *(const uint4*)(Vb + (size_t)(kb * 64 + key) * D_ + dc);
      #pragma unroll
      for (int j = 0; j < 8; ++j) VT[(dc + j) * 72 + key] = u.s[j];
    }
    __syncthreads();

    // ---- scores S = Q K^T ----
    v8f sc[4];
    #pragma unroll
    for (int t = 0; t < 4; ++t) {
      sc[t] = zero;
      #pragma unroll
      for (int ks = 0; ks < 2; ++ks) {
        // B layout: lane col n = l16; K = ks*32 + hi*8..+7 and +16..+23 in [s][d]
        const unsigned short* kp = Kb + (size_t)(kb * 64 + t * 16 + l16) * D_ + ks * 32 + hi * 8;
        v16bf kB = ld_frag(kp, kp + 16);
        sc[t] = __builtin_amdgcn_wmma_f32_16x16x32_bf16(
            false, qA[ks], false, kB, (short)0, sc[t], false, false);
      }
    }

    // ---- scale + mask penalty ----
    #pragma unroll
    for (int t = 0; t < 4; ++t) {
      const int key_in = t * 16 + l16;
      const int key_g  = kb * 64 + key_in;
      float addT = 0.f, tbk = 0.f;
      if (mode == 0) addT = (1.f - to_mask[(size_t)b * S_ + key_g]) * PEN;
      if (mode == 2) tbk  = to_blocked[((size_t)b * NB_ + kb) * BLK_ + key_in];
      #pragma unroll
      for (int v = 0; v < 8; ++v) {
        float add;
        if (mode == 0) add = addT;
        else if (mode == 1) {
          int qrow = w * 16 + hi * 8 + v;
          add = (1.f - band_mask[(((size_t)b * (NB_ - 4) + (qb - 2)) * BLK_ + qrow) * (3 * BLK_)
                                 + rel * BLK_ + key_in]) * PEN;
        } else {
          add = (1.f - fbv[v] * tbk) * PEN;
        }
        sc[t][v] = sc[t][v] * 0.125f + add;   // 1/sqrt(64)
      }
    }

    // ---- online softmax update (row reductions across 16 lanes) ----
    float bm8[8];
    #pragma unroll
    for (int v = 0; v < 8; ++v)
      bm8[v] = fmaxf(fmaxf(sc[0][v], sc[1][v]), fmaxf(sc[2][v], sc[3][v]));
    #pragma unroll
    for (int off = 1; off < 16; off <<= 1) {
      #pragma unroll
      for (int v = 0; v < 8; ++v)
        bm8[v] = fmaxf(bm8[v], __shfl_xor(bm8[v], off, 32));
    }

    float ps[8];
    #pragma unroll
    for (int v = 0; v < 8; ++v) {
      float mn    = fmaxf(mrow[v], bm8[v]);
      float scale = __expf(mrow[v] - mn);
      mrow[v] = mn;
      lrow[v] *= scale;
      #pragma unroll
      for (int t = 0; t < 4; ++t) accO[t][v] *= scale;
      float pa = 0.f;
      #pragma unroll
      for (int t = 0; t < 4; ++t) {
        float p = __expf(sc[t][v] - mn);
        pa += p;
        Pl[w][(hi * 8 + v) * 72 + t * 16 + l16] = f2bf(p);   // C -> A transpose via LDS
      }
      ps[v] = pa;
    }
    #pragma unroll
    for (int off = 1; off < 16; off <<= 1) {
      #pragma unroll
      for (int v = 0; v < 8; ++v) ps[v] += __shfl_xor(ps[v], off, 32);
    }
    #pragma unroll
    for (int v = 0; v < 8; ++v) lrow[v] += ps[v];

    __syncthreads();   // make P tile visible, keep waves in lockstep before VT reuse

    // ---- O += P V ----
    #pragma unroll
    for (int ks = 0; ks < 2; ++ks) {
      const unsigned short* pp = &Pl[w][l16 * 72 + ks * 32 + hi * 8];
      v16bf pA = ld_frag(pp, pp + 16);
      #pragma unroll
      for (int t = 0; t < 4; ++t) {
        const unsigned short* vp = &VT[(t * 16 + l16) * 72 + ks * 32 + hi * 8];
        v16bf vB = ld_frag(vp, vp + 16);
        accO[t] = __builtin_amdgcn_wmma_f32_16x16x32_bf16(
            false, pA, false, vB, (short)0, accO[t], false, false);
      }
    }
  }

  // ---- epilogue: normalize, apply from_mask, scatter to (B,S,H) ----
  #pragma unroll
  for (int v = 0; v < 8; ++v) {
    const int qrow = w * 16 + hi * 8 + v;
    const int sg   = qb * 64 + qrow;
    const float fm  = from_mask[(size_t)b * S_ + sg];
    const float inv = fm / lrow[v];
    size_t ob = ((size_t)b * S_ + sg) * H_ + (size_t)h * 64;
    #pragma unroll
    for (int t = 0; t < 4; ++t) out[ob + t * 16 + l16] = accO[t][v] * inv;
  }
}

// ---------------------------------------------------------------------------
extern "C" void kernel_launch(void* const* d_in, const int* in_sizes, int n_in,
                              void* d_out, int out_size, void* d_ws, size_t ws_size,
                              hipStream_t stream) {
  const float* X     = (const float*)d_in[0];
  const float* Wq    = (const float*)d_in[1];
  const float* Wk    = (const float*)d_in[2];
  const float* Wv    = (const float*)d_in[3];
  const float* band  = (const float*)d_in[4];
  const float* fmask = (const float*)d_in[5];
  const float* tmask = (const float*)d_in[6];
  const float* fb    = (const float*)d_in[7];
  const float* tb    = (const float*)d_in[8];
  const long long* ra = (const long long*)d_in[9];   // int64 rand_attn

  const size_t per = (size_t)B_ * NH_ * S_ * D_;     // 6,291,456 bf16 elems
  const size_t NX  = (size_t)B_ * S_ * H_;           // 6,291,456
  const size_t NW  = (size_t)3 * H_ * H_;            // 1,769,472

  unsigned short* Qw  = (unsigned short*)d_ws;
  unsigned short* Kw  = Qw + per;
  unsigned short* Vw  = Kw + per;
  unsigned short* Xb  = Vw + per;
  unsigned short* WTb = Xb + NX;

  const int nconv = (int)((NX + NW + 255) / 256);
  convert_kernel<<<nconv, 256, 0, stream>>>(X, Wq, Wk, Wv, Xb, WTb);

  qkv_proj_kernel<<<dim3((B_ * S_) / 64, NH_, 3), 128, 0, stream>>>(
      Xb, WTb, Qw, Kw, Vw);

  bigbird_attn_kernel<<<dim3(NB_, NH_, B_), 128, 0, stream>>>(
      Qw, Kw, Vw, band, fmask, tmask, fb, tb, ra, (float*)d_out);
}
